// StructureAwareTransformer_35450660061796
// MI455X (gfx1250) — compile-verified, matrix operand order checked by reading, not run on hardware
//
#include <hip/hip_runtime.h>
#include <hip/hip_bf16.h>

// ---------------------------------------------------------------------------
// StructureAwareTransformer GNN message passing for gfx1250 (MI455X).
// Strategy: bf16 WMMA (v_wmma_f32_16x16x32_bf16) GEMMs with f32 accumulate,
// fused gather -> MLP -> bias/ReLU -> residual -> scatter(atomic f32) kernels.
// Weights pre-permuted into WMMA-B-fragment-native layout (2x b128 per lane).
// ---------------------------------------------------------------------------

typedef __bf16 bf16_t;
typedef __attribute__((ext_vector_type(16))) __bf16 v16bf;
typedef __attribute__((ext_vector_type(8)))  float  v8f;
typedef __attribute__((ext_vector_type(4)))  unsigned int v4u;

#define GN  50000      // nodes
#define GE  400000     // edges
#define GD  128        // d_model
#define GK  2          // hops

#define TM       64    // rows per block
#define ASTRIDE  392   // bf16 row stride of A tile (384 + 8 pad), 784B = 16B-aligned
#define HSTRIDE  136   // bf16 row stride of hidden tile (128 + 8 pad), 272B aligned

union Frag { v16bf v; v4u q[2]; };

__device__ __forceinline__ v8f wmma_bf16(v16bf a, v16bf b, v8f c) {
  // (neg_a, A, neg_b, B, c_mod, C, reuse_a, reuse_b)
  return __builtin_amdgcn_wmma_f32_16x16x32_bf16(false, a, false, b,
                                                 (short)0, c, false, false);
}

// A fragment from an LDS row-major bf16 tile. p points at (row, k0 + khalf).
__device__ __forceinline__ v16bf frag_from_lds(const bf16_t* p) {
  Frag f;
  f.q[0] = *(const v4u*)(p);        // K = base + 0..7
  f.q[1] = *(const v4u*)(p + 16);   // K = base + 16..23
  return f.v;
}

// B fragment from pre-permuted global weights: layout [ktile][ntile=8][lane][16]
__device__ __forceinline__ v16bf frag_from_w(const bf16_t* wp, int kt, int nt, int lane) {
  const bf16_t* p = wp + ((((kt << 3) + nt) << 5) + lane) * 16; // 32B per lane
  Frag f;
  f.q[0] = ((const v4u*)p)[0];
  f.q[1] = ((const v4u*)p)[1];
  return f.v;
}

// Run the 2-layer MLP on a 64x384 bf16 tile in LDS (shA), produce per-wave
// accumulators acc2[4] (wave owns column tile == wave id). shA is clobbered.
__device__ __forceinline__ void mlp_64rows(bf16_t* shA,
                                           const bf16_t* W1p, const bf16_t* W2p,
                                           const float* b1, int wave, int lane,
                                           v8f acc2[4]) {
  const int mlane = lane & 15;
  const int khalf = (lane >> 4) << 3;

  v8f acc1[4];
  #pragma unroll
  for (int m = 0; m < 4; ++m) acc1[m] = (v8f){0.f,0.f,0.f,0.f,0.f,0.f,0.f,0.f};

  // Layer 1: A[64x384] x W1[384x128], K tiles = 12
  for (int kt = 0; kt < 12; ++kt) {
    v16bf bf = frag_from_w(W1p, kt, wave, lane);
    const bf16_t* abase = shA + mlane * ASTRIDE + (kt << 5) + khalf;
    #pragma unroll
    for (int m = 0; m < 4; ++m) {
      v16bf af = frag_from_lds(abase + m * 16 * ASTRIDE);
      acc1[m] = wmma_bf16(af, bf, acc1[m]);
    }
  }

  // bias + ReLU, hidden tile back into LDS (reuse shA) as bf16
  const int col = (wave << 4) + mlane;
  const float b1v = b1[col];
  __syncthreads();                       // all layer-1 reads of shA done
  bf16_t* shH = shA;
  #pragma unroll
  for (int m = 0; m < 4; ++m) {
    #pragma unroll
    for (int r = 0; r < 8; ++r) {
      int row = (m << 4) + r + khalf;
      float v = acc1[m][r] + b1v;
      v = v > 0.f ? v : 0.f;
      shH[row * HSTRIDE + col] = (bf16_t)v;
    }
  }
  __syncthreads();

  // Layer 2: H[64x128] x W2[128x128], K tiles = 4
  #pragma unroll
  for (int m = 0; m < 4; ++m) acc2[m] = (v8f){0.f,0.f,0.f,0.f,0.f,0.f,0.f,0.f};
  for (int kt = 0; kt < 4; ++kt) {
    v16bf bf = frag_from_w(W2p, kt, wave, lane);
    const bf16_t* hbase = shH + mlane * HSTRIDE + (kt << 5) + khalf;
    #pragma unroll
    for (int m = 0; m < 4; ++m) {
      v16bf af = frag_from_lds(hbase + m * 16 * HSTRIDE);
      acc2[m] = wmma_bf16(af, bf, acc2[m]);
    }
  }
}

// ---------------------------------------------------------------------------
// Edge kernel: new_e = MLP(concat(edges, nodes[s], nodes[r]));
// edges += new_e; atomic scatter-add new_e into sent[s], recv[r].
// ---------------------------------------------------------------------------
__global__ void __launch_bounds__(256)
k_edge(float* edges, const float* __restrict__ nodes,
       const int* __restrict__ senders, const int* __restrict__ receivers,
       const bf16_t* __restrict__ W1p, const bf16_t* __restrict__ W2p,
       const float* __restrict__ b1, const float* __restrict__ b2,
       float* sent, float* recv) {
  __shared__ __align__(16) bf16_t shA[TM * ASTRIDE];
  __shared__ int shS[TM], shR[TM];

  const int tid  = threadIdx.x;
  const int lane = tid & 31;
  const int wave = tid >> 5;
  const int e0   = blockIdx.x * TM;

  if (tid < TM) { shS[tid] = senders[e0 + tid]; shR[tid] = receivers[e0 + tid]; }
  __syncthreads();

  // Build A tile: [edge | node[s] | node[r]] -> bf16
  for (int i = tid; i < TM * 384; i += 256) {
    int row = i / 384, c = i - row * 384;
    float v;
    if (c < 128)      v = edges[(e0 + row) * GD + c];
    else if (c < 256) v = nodes[shS[row] * GD + (c - 128)];
    else              v = nodes[shR[row] * GD + (c - 256)];
    shA[row * ASTRIDE + c] = (bf16_t)v;
  }
  __syncthreads();

  v8f acc2[4];
  mlp_64rows(shA, W1p, W2p, b1, wave, lane, acc2);

  const int col   = (wave << 4) + (lane & 15);
  const int khalf = (lane >> 4) << 3;
  const float b2v = b2[col];
  #pragma unroll
  for (int m = 0; m < 4; ++m) {
    #pragma unroll
    for (int r = 0; r < 8; ++r) {
      int row = (m << 4) + r + khalf;
      float v = acc2[m][r] + b2v;
      v = v > 0.f ? v : 0.f;
      int gi = (e0 + row) * GD + col;
      edges[gi] = edges[gi] + v;                      // residual, row owned by us
      unsafeAtomicAdd(&sent[shS[row] * GD + col], v); // global_atomic_add_f32
      unsafeAtomicAdd(&recv[shR[row] * GD + col], v);
    }
  }
}

// ---------------------------------------------------------------------------
// Node kernel: nodes += MLP(concat(nodes, sent, recv))
// ---------------------------------------------------------------------------
__global__ void __launch_bounds__(256)
k_node(float* nodes, const float* __restrict__ sent, const float* __restrict__ recv,
       const bf16_t* __restrict__ W1p, const bf16_t* __restrict__ W2p,
       const float* __restrict__ b1, const float* __restrict__ b2, int nN) {
  __shared__ __align__(16) bf16_t shA[TM * ASTRIDE];

  const int tid  = threadIdx.x;
  const int lane = tid & 31;
  const int wave = tid >> 5;
  const int n0   = blockIdx.x * TM;
  const int rows = (nN - n0) < TM ? (nN - n0) : TM;

  for (int i = tid; i < TM * 384; i += 256) {
    int row = i / 384, c = i - row * 384;
    float v = 0.f;
    if (row < rows) {
      int n = n0 + row;
      if (c < 128)      v = nodes[n * GD + c];
      else if (c < 256) v = sent[n * GD + (c - 128)];
      else              v = recv[n * GD + (c - 256)];
    }
    shA[row * ASTRIDE + c] = (bf16_t)v;
  }
  __syncthreads();

  v8f acc2[4];
  mlp_64rows(shA, W1p, W2p, b1, wave, lane, acc2);

  const int col   = (wave << 4) + (lane & 15);
  const int khalf = (lane >> 4) << 3;
  const float b2v = b2[col];
  #pragma unroll
  for (int m = 0; m < 4; ++m) {
    #pragma unroll
    for (int r = 0; r < 8; ++r) {
      int row = (m << 4) + r + khalf;
      if (row < rows) {
        float v = acc2[m][r] + b2v;
        v = v > 0.f ? v : 0.f;
        int gi = (n0 + row) * GD + col;
        nodes[gi] = nodes[gi] + v;
      }
    }
  }
}

// ---------------------------------------------------------------------------
// Permute a row-major f32 weight [ktiles*32, 128] into fragment-native bf16:
// out[((kt*8 + nt)*32 + lane)*16 + e] = W[kt*32 + kmap(lane,e)][nt*16 + lane%16]
// kmap(l,e) = e%8 + (e/8)*16 + (l/16)*8   (ISA 7.12.2, 16-bit layout)
// ---------------------------------------------------------------------------
__global__ void __launch_bounds__(256)
k_permw(const float* __restrict__ W, bf16_t* __restrict__ Wp, int total) {
  int idx = blockIdx.x * 256 + threadIdx.x;
  if (idx >= total) return;
  int e  = idx & 15;
  int l  = (idx >> 4) & 31;
  int nt = (idx >> 9) & 7;
  int kt = idx >> 12;
  int k  = (kt << 5) + (e & 7) + ((e >> 3) << 4) + ((l >> 4) << 3);
  int n  = (nt << 4) + (l & 15);
  Wp[idx] = (bf16_t)W[k * GD + n];
}

__global__ void __launch_bounds__(256)
k_zero(float* p, int n) {
  int i = blockIdx.x * 256 + threadIdx.x;
  int stride = gridDim.x * 256;
  for (; i < n; i += stride) p[i] = 0.f;
}

extern "C" void kernel_launch(void* const* d_in, const int* in_sizes, int n_in,
                              void* d_out, int out_size, void* d_ws, size_t ws_size,
                              hipStream_t stream) {
  const float* nodes_in  = (const float*)d_in[0];
  const float* edges_in  = (const float*)d_in[1];
  const float* eW1 = (const float*)d_in[2];
  const float* eb1 = (const float*)d_in[3];
  const float* eW2 = (const float*)d_in[4];
  const float* eb2 = (const float*)d_in[5];
  const float* nW1 = (const float*)d_in[6];
  const float* nb1 = (const float*)d_in[7];
  const float* nW2 = (const float*)d_in[8];
  const float* nb2 = (const float*)d_in[9];
  const int* senders   = (const int*)d_in[10];
  const int* receivers = (const int*)d_in[11];

  // d_out = [nodes (N*D) | edges (E*D)] -- used in place as working buffers.
  float* nodes = (float*)d_out;
  float* edges = (float*)d_out + (size_t)GN * GD;

  // Workspace: sent, recv accumulators + permuted bf16 weights.
  float* sent = (float*)d_ws;
  float* recv = sent + (size_t)GN * GD;
  bf16_t* wp   = (bf16_t*)(recv + (size_t)GN * GD);
  const int W1E = 384 * GD;   // 49152 elements per hop
  const int W2E = GD * GD;    // 16384 elements per hop
  bf16_t* eW1p = wp;
  bf16_t* eW2p = eW1p + (size_t)GK * W1E;
  bf16_t* nW1p = eW2p + (size_t)GK * W2E;
  bf16_t* nW2p = nW1p + (size_t)GK * W1E;

  hipMemcpyAsync(nodes, nodes_in, (size_t)GN * GD * sizeof(float),
                 hipMemcpyDeviceToDevice, stream);
  hipMemcpyAsync(edges, edges_in, (size_t)GE * GD * sizeof(float),
                 hipMemcpyDeviceToDevice, stream);

  for (int h = 0; h < GK; ++h) {
    k_permw<<<(W1E + 255) / 256, 256, 0, stream>>>(eW1 + (size_t)h * W1E, eW1p + (size_t)h * W1E, W1E);
    k_permw<<<(W2E + 255) / 256, 256, 0, stream>>>(eW2 + (size_t)h * W2E, eW2p + (size_t)h * W2E, W2E);
    k_permw<<<(W1E + 255) / 256, 256, 0, stream>>>(nW1 + (size_t)h * W1E, nW1p + (size_t)h * W1E, W1E);
    k_permw<<<(W2E + 255) / 256, 256, 0, stream>>>(nW2 + (size_t)h * W2E, nW2p + (size_t)h * W2E, W2E);
  }

  const int aggN = 2 * GN * GD;  // sent+recv contiguous
  for (int h = 0; h < GK; ++h) {
    k_zero<<<2048, 256, 0, stream>>>(sent, aggN);
    k_edge<<<GE / TM, 256, 0, stream>>>(edges, nodes, senders, receivers,
                                        eW1p + (size_t)h * W1E, eW2p + (size_t)h * W2E,
                                        eb1 + h * GD, eb2 + h * GD, sent, recv);
    k_node<<<(GN + TM - 1) / TM, 256, 0, stream>>>(nodes, sent, recv,
                                                   nW1p + (size_t)h * W1E, nW2p + (size_t)h * W2E,
                                                   nb1 + h * GD, nb2 + h * GD, GN);
  }
}
